// Transformer_Block_21577915695340
// MI455X (gfx1250) — matrix operand order repaired and checked by reading, not compiled, not run on hardware
//
#include <hip/hip_runtime.h>
#include <math.h>

// ---------------------------------------------------------------------------
// Types for WMMA fragments (wave32, v_wmma_f32_16x16x32_bf16)
// ---------------------------------------------------------------------------
typedef __attribute__((ext_vector_type(4)))  int             v4i;
typedef __attribute__((ext_vector_type(4)))  unsigned        u32x4;
typedef __attribute__((ext_vector_type(8)))  unsigned        u32x8;
typedef __attribute__((ext_vector_type(8)))  unsigned short  u16x8;
typedef __attribute__((ext_vector_type(8)))  float           v8f;
typedef __attribute__((ext_vector_type(16))) __bf16          v16bf;

__device__ __forceinline__ unsigned short f2bf(float f) {
    unsigned u = __builtin_bit_cast(unsigned, f);
    u += 0x7FFFu + ((u >> 16) & 1u);       // round-to-nearest-even
    return (unsigned short)(u >> 16);
}

__device__ __forceinline__ u32x8 frag_combine(u32x4 lo, u32x4 hi) {
    u32x8 r;
    r[0] = lo[0]; r[1] = lo[1]; r[2] = lo[2]; r[3] = lo[3];
    r[4] = hi[0]; r[5] = hi[1]; r[6] = hi[2]; r[7] = hi[3];
    return r;
}

__device__ __forceinline__ v8f wmma_bf16(u32x8 a, u32x8 b, v8f c) {
    return __builtin_amdgcn_wmma_f32_16x16x32_bf16(
        /*neg_a=*/false, __builtin_bit_cast(v16bf, a),
        /*neg_b=*/false, __builtin_bit_cast(v16bf, b),
        /*c_mod=*/(short)0, c, /*reuse_a=*/false, /*reuse_b=*/false);
}

// ---------------------------------------------------------------------------
// CDNA5 async global->LDS copy path (ASYNCcnt), with sync fallback
// The builtin's pointer params are int4* (Sema rewrites the address space to
// match the arguments, but the pointee type must be the 4 x i32 vector).
// ---------------------------------------------------------------------------
#if __has_builtin(__builtin_amdgcn_global_load_async_to_lds_b128) && \
    __has_builtin(__builtin_amdgcn_s_wait_asynccnt)
#define ASYNC_CP16(g, l)                                              \
    __builtin_amdgcn_global_load_async_to_lds_b128(                   \
        (v4i*)(g), (v4i*)(l), 0, 0)
#define ASYNC_WAIT() __builtin_amdgcn_s_wait_asynccnt(0)
#else
#define ASYNC_CP16(g, l) (*(u32x4*)(l) = *(const u32x4*)(g))
#define ASYNC_WAIT() ((void)0)
#endif

// ---------------------------------------------------------------------------
// Problem constants
// ---------------------------------------------------------------------------
#define NB   16
#define NTOK 577
#define NC   768
#define NH   12
#define NDH  64
#define NFF  3072
#define MROWS (NB * NTOK)     // 9232

// ---------------------------------------------------------------------------
// Weight cast + transpose: fp32 [K][N] -> bf16 [N][K]
// ---------------------------------------------------------------------------
__global__ __launch_bounds__(256) void castT_kernel(
    const float* __restrict__ src, unsigned short* __restrict__ dst,
    int K, int N)
{
    size_t total = (size_t)K * N;
    for (size_t i = (size_t)blockIdx.x * 256 + threadIdx.x; i < total;
         i += (size_t)gridDim.x * 256) {
        size_t k = i / N, n = i - k * N;
        dst[n * (size_t)K + k] = f2bf(src[i]);
    }
}

// ---------------------------------------------------------------------------
// LayerNorm: fp32 [M][768] -> bf16 [M][768]
// ---------------------------------------------------------------------------
__global__ __launch_bounds__(256) void ln_kernel(
    const float* __restrict__ x, const float* __restrict__ g,
    const float* __restrict__ b, unsigned short* __restrict__ out)
{
    __shared__ float wsum[8], wsq[8];
    const int row = blockIdx.x, tid = threadIdx.x;
    const float* xr = x + (size_t)row * NC;
    float v0 = xr[tid], v1 = xr[tid + 256], v2 = xr[tid + 512];
    float s = v0 + v1 + v2;
    float q = v0 * v0 + v1 * v1 + v2 * v2;
    for (int m = 1; m < 32; m <<= 1) {
        s += __shfl_xor(s, m);
        q += __shfl_xor(q, m);
    }
    if ((tid & 31) == 0) { wsum[tid >> 5] = s; wsq[tid >> 5] = q; }
    __syncthreads();
    float ts = 0.f, tq = 0.f;
    for (int i = 0; i < 8; i++) { ts += wsum[i]; tq += wsq[i]; }
    float mean = ts * (1.0f / NC);
    float var  = tq * (1.0f / NC) - mean * mean;
    float rstd = rsqrtf(var + 1e-5f);
    unsigned short* o = out + (size_t)row * NC;
    o[tid]       = f2bf((v0 - mean) * rstd * g[tid]       + b[tid]);
    o[tid + 256] = f2bf((v1 - mean) * rstd * g[tid + 256] + b[tid + 256]);
    o[tid + 512] = f2bf((v2 - mean) * rstd * g[tid + 512] + b[tid + 512]);
}

// ---------------------------------------------------------------------------
// Tiled WMMA GEMM: C[m][n] = sum_k A[m][k] * Wt[n][k] (+ bias, epilogue)
//   block tile 128x128, 8 waves, wave tile 32x64, K-step 32,
//   double-buffered LDS with async global->LDS copies.
//   MODE 0: bf16 out, head-split [B,H,N,DH] layout (QKV)
//   MODE 1: fp32 out = acc + bias + resid         (proj / final)
//   MODE 2: bf16 out = gelu(acc + bias)           (MLP up)
// ---------------------------------------------------------------------------
#define LDT 40   // 32 + 8 halfword padding

#define STAGE_TILES(k0_, buf_)                                                 \
    do {                                                                       \
        for (int s_ = tid; s_ < 512; s_ += 256) {                              \
            int row_ = s_ >> 2, kq_ = (s_ & 3) * 8;                            \
            int ar_ = tm * 128 + row_; if (ar_ > M - 1) ar_ = M - 1;           \
            int nr_ = tn * 128 + row_;                                         \
            ASYNC_CP16(A  + (size_t)ar_ * K + (k0_) + kq_,                     \
                       &Asm[buf_][row_ * LDT + kq_]);                          \
            ASYNC_CP16(Wt + (size_t)nr_ * K + (k0_) + kq_,                     \
                       &Bsm[buf_][row_ * LDT + kq_]);                          \
        }                                                                      \
    } while (0)

template <int MODE>
__global__ __launch_bounds__(256) void gemm_kernel(
    const unsigned short* __restrict__ A, const unsigned short* __restrict__ Wt,
    const float* __restrict__ bias, const float* __restrict__ resid,
    void* __restrict__ outp, int M, int K, int Nout)
{
    __shared__ unsigned short Asm[2][128 * LDT];
    __shared__ unsigned short Bsm[2][128 * LDT];
    const int tid  = threadIdx.x;
    const int lane = tid & 31, wid = tid >> 5;
    const int wm = wid >> 1, wn = wid & 1;
    const int tm = blockIdx.y, tn = blockIdx.x;
    const int half = lane >> 4;
    const int l16  = lane & 15;

    v8f zero8 = {0.f, 0.f, 0.f, 0.f, 0.f, 0.f, 0.f, 0.f};
    v8f acc[2][4];
    for (int i = 0; i < 2; i++)
        for (int j = 0; j < 4; j++) acc[i][j] = zero8;

    const int abase = half ? 8 : 0;    // A frag K start within 32-tile
    const int bbase = half ? 8 : 0;    // B frag K start within 32-tile

    // prologue: stage first tile
    STAGE_TILES(0, 0);
    ASYNC_WAIT();
    __syncthreads();

    const int nk32 = K >> 5;
    for (int it = 0; it < nk32; ++it) {
        const int buf = it & 1;
        // issue async copy of the next tile while computing on the current
        if (it + 1 < nk32) STAGE_TILES((it + 1) * 32, buf ^ 1);

        u32x8 af[2], bf[4];
#pragma unroll
        for (int fm = 0; fm < 2; fm++) {
            int r = wm * 32 + fm * 16 + l16;
            af[fm] = frag_combine(*(const u32x4*)(&Asm[buf][r * LDT + abase]),
                                  *(const u32x4*)(&Asm[buf][r * LDT + abase + 16]));
        }
#pragma unroll
        for (int fn = 0; fn < 4; fn++) {
            int r = wn * 64 + fn * 16 + l16;
            bf[fn] = frag_combine(*(const u32x4*)(&Bsm[buf][r * LDT + bbase]),
                                  *(const u32x4*)(&Bsm[buf][r * LDT + bbase + 16]));
        }
#pragma unroll
        for (int fm = 0; fm < 2; fm++)
#pragma unroll
            for (int fn = 0; fn < 4; fn++)
                acc[fm][fn] = wmma_bf16(af[fm], bf[fn], acc[fm][fn]);

        ASYNC_WAIT();
        __syncthreads();
    }

    // ---- epilogue ----
    for (int fm = 0; fm < 2; fm++) {
        for (int fn = 0; fn < 4; fn++) {
            int col = tn * 128 + wn * 64 + fn * 16 + l16;
            float bv = bias[col];
            for (int r = 0; r < 8; r++) {
                int row = tm * 128 + wm * 32 + fm * 16 + half * 8 + r;
                if (row >= M) continue;
                float v = acc[fm][fn][r] + bv;
                if constexpr (MODE == 0) {
                    int bb = row / NTOK, tok = row - bb * NTOK;
                    int hh = col >> 6, dh = col & 63;
                    ((unsigned short*)outp)[(((size_t)bb * NH + hh) * NTOK + tok) * NDH + dh] = f2bf(v);
                } else if constexpr (MODE == 1) {
                    size_t idx = (size_t)row * Nout + col;
                    ((float*)outp)[idx] = v + resid[idx];
                } else {
                    float gl = 0.5f * v * (1.0f + erff(v * 0.70710678118f));
                    ((unsigned short*)outp)[(size_t)row * Nout + col] = f2bf(gl);
                }
            }
        }
    }
}

// ---------------------------------------------------------------------------
// Flash attention: one (b,h) pair per blockIdx.y, 128 query rows per block,
// each wave owns 16 query rows; keys streamed in blocks of 32.
// Q/K/V are bf16 in [B,H,N,DH] layout; output bf16 in [B*N][C].
// ---------------------------------------------------------------------------
__global__ __launch_bounds__(256) void attn_kernel(
    const unsigned short* __restrict__ Q, const unsigned short* __restrict__ Km,
    const unsigned short* __restrict__ Vm, unsigned short* __restrict__ Ob)
{
    __shared__ unsigned short Vt[64 * LDT];        // V^T tile: [dh][key]
    __shared__ unsigned short St[8 * 16 * LDT];    // per-wave P staging
    const int tid = threadIdx.x, lane = tid & 31, wid = tid >> 5;
    const int l16 = lane & 15, half = lane >> 4;
    const int bh = blockIdx.y;
    const int b  = bh / NH, h = bh % NH;
    const int q0 = blockIdx.x * 128 + wid * 16;
    const int abase = half ? 8 : 0;
    const int bbase = half ? 8 : 0;

    // ---- Q fragments (A-layout, dh 0..31 and 32..63), loaded once ----
    int qg = q0 + l16; if (qg > NTOK - 1) qg = NTOK - 1;
    const unsigned short* qptr = Q + ((size_t)bh * NTOK + qg) * NDH;
    u32x8 A0 = frag_combine(*(const u32x4*)(qptr + abase),
                            *(const u32x4*)(qptr + abase + 16));
    u32x8 A1 = frag_combine(*(const u32x4*)(qptr + abase + 32),
                            *(const u32x4*)(qptr + abase + 48));

    v8f zero8 = {0.f, 0.f, 0.f, 0.f, 0.f, 0.f, 0.f, 0.f};
    float m[8], lsum[8];
    v8f Oa[4];
    for (int r = 0; r < 8; r++) { m[r] = -1e30f; lsum[r] = 0.f; }
    for (int d = 0; d < 4; d++) Oa[d] = zero8;

    unsigned short* Sw = &St[wid * 16 * LDT];
    const int nkb = (NTOK + 31) / 32;   // 19

    for (int kb = 0; kb < nkb; kb++) {
        __syncthreads();   // prior iteration's Vt reads complete
        // ---- cooperative V^T staging ----
        {
            int key_l = tid >> 3;
            int dq    = (tid & 7) * 8;
            int keyg  = kb * 32 + key_l; if (keyg > NTOK - 1) keyg = NTOK - 1;
            u32x4 vv = *(const u32x4*)(Vm + ((size_t)bh * NTOK + keyg) * NDH + dq);
            u16x8 vs = __builtin_bit_cast(u16x8, vv);
            for (int j = 0; j < 8; j++) Vt[(dq + j) * LDT + key_l] = vs[j];
        }
        __syncthreads();

        // ---- scores: S = Q K^T / sqrt(DH), two 16x16 fragments ----
        v8f  s[2];
        bool valid[2];
        for (int j = 0; j < 2; j++) {
            int kbase = kb * 32 + j * 16;
            int keyg  = kbase + l16;
            valid[j]  = (keyg < NTOK);
            if (keyg > NTOK - 1) keyg = NTOK - 1;
            const unsigned short* kptr = Km + ((size_t)bh * NTOK + keyg) * NDH;
            u32x8 B0 = frag_combine(*(const u32x4*)(kptr + bbase),
                                    *(const u32x4*)(kptr + bbase + 16));
            u32x8 B1 = frag_combine(*(const u32x4*)(kptr + 32 + bbase),
                                    *(const u32x4*)(kptr + 32 + bbase + 16));
            v8f t = zero8;
            t = wmma_bf16(A0, B0, t);
            t = wmma_bf16(A1, B1, t);
            s[j] = t;
        }

        // ---- online softmax update ----
        for (int r = 0; r < 8; r++) {
            float v0 = valid[0] ? s[0][r] * 0.125f : -1e30f;
            float v1 = valid[1] ? s[1][r] * 0.125f : -1e30f;
            float mx = fmaxf(v0, v1);
            mx = fmaxf(mx, __shfl_xor(mx, 1));
            mx = fmaxf(mx, __shfl_xor(mx, 2));
            mx = fmaxf(mx, __shfl_xor(mx, 4));
            mx = fmaxf(mx, __shfl_xor(mx, 8));
            float mn    = fmaxf(m[r], mx);
            float alpha = __expf(m[r] - mn);
            m[r] = mn;
            float p0 = valid[0] ? __expf(v0 - mn) : 0.f;
            float p1 = valid[1] ? __expf(v1 - mn) : 0.f;
            lsum[r] = lsum[r] * alpha + p0 + p1;
            for (int d = 0; d < 4; d++) Oa[d][r] *= alpha;
            int srow = half * 8 + r;
            Sw[srow * LDT + l16]      = f2bf(p0);
            Sw[srow * LDT + 16 + l16] = f2bf(p1);
        }

        // ---- O += P V (A-layout P from LDS, B-layout V^T from LDS) ----
        u32x8 Pa = frag_combine(*(const u32x4*)(&Sw[l16 * LDT + abase]),
                                *(const u32x4*)(&Sw[l16 * LDT + abase + 16]));
        for (int d = 0; d < 4; d++) {
            int n = d * 16 + l16;
            u32x8 Vb = frag_combine(*(const u32x4*)(&Vt[n * LDT + bbase]),
                                    *(const u32x4*)(&Vt[n * LDT + bbase + 16]));
            Oa[d] = wmma_bf16(Pa, Vb, Oa[d]);
        }
    }

    // ---- finalize: divide by row sums, write heads concatenated ----
    float inv[8];
    for (int r = 0; r < 8; r++) {
        float lv = lsum[r];
        lv += __shfl_xor(lv, 1);
        lv += __shfl_xor(lv, 2);
        lv += __shfl_xor(lv, 4);
        lv += __shfl_xor(lv, 8);
        inv[r] = 1.0f / lv;
    }
    for (int d = 0; d < 4; d++) {
        int c = h * NDH + d * 16 + l16;
        for (int r = 0; r < 8; r++) {
            int qr = q0 + half * 8 + r;
            if (qr < NTOK)
                Ob[((size_t)b * NTOK + qr) * NC + c] = f2bf(Oa[d][r] * inv[r]);
        }
    }
}

// ---------------------------------------------------------------------------
// Host-side launch
// ---------------------------------------------------------------------------
extern "C" void kernel_launch(void* const* d_in, const int* in_sizes, int n_in,
                              void* d_out, int out_size, void* d_ws, size_t ws_size,
                              hipStream_t stream) {
    (void)in_sizes; (void)n_in; (void)out_size; (void)ws_size;
    const float* x    = (const float*)d_in[0];
    const float* ln1g = (const float*)d_in[2];
    const float* ln1b = (const float*)d_in[3];
    const float* wq = (const float*)d_in[4];  const float* bq = (const float*)d_in[5];
    const float* wk = (const float*)d_in[6];  const float* bk = (const float*)d_in[7];
    const float* wv = (const float*)d_in[8];  const float* bv = (const float*)d_in[9];
    const float* wo = (const float*)d_in[10]; const float* bo = (const float*)d_in[11];
    const float* ln2g = (const float*)d_in[12]; const float* ln2b = (const float*)d_in[13];
    const float* w1 = (const float*)d_in[14]; const float* b1 = (const float*)d_in[15];
    const float* w2 = (const float*)d_in[16]; const float* b2 = (const float*)d_in[17];

    char* p = (char*)d_ws;
    auto alloc = [&](size_t bytes) {
        void* r = (void*)p;
        p += (bytes + 255) & ~(size_t)255;
        return r;
    };
    unsigned short* wqT = (unsigned short*)alloc((size_t)NC * NC * 2);
    unsigned short* wkT = (unsigned short*)alloc((size_t)NC * NC * 2);
    unsigned short* wvT = (unsigned short*)alloc((size_t)NC * NC * 2);
    unsigned short* woT = (unsigned short*)alloc((size_t)NC * NC * 2);
    unsigned short* w1T = (unsigned short*)alloc((size_t)NC * NFF * 2);
    unsigned short* w2T = (unsigned short*)alloc((size_t)NFF * NC * 2);
    unsigned short* hbuf = (unsigned short*)alloc((size_t)MROWS * NC * 2);
    unsigned short* Qb   = (unsigned short*)alloc((size_t)MROWS * NC * 2);
    unsigned short* Kb   = (unsigned short*)alloc((size_t)MROWS * NC * 2);
    unsigned short* Vb   = (unsigned short*)alloc((size_t)MROWS * NC * 2);
    unsigned short* AOb  = (unsigned short*)alloc((size_t)MROWS * NC * 2);
    float*          x1   = (float*)alloc((size_t)MROWS * NC * 4);
    unsigned short* h2   = (unsigned short*)alloc((size_t)MROWS * NC * 2);
    unsigned short* mid  = (unsigned short*)alloc((size_t)MROWS * NFF * 2);

    const int tilesM = (MROWS + 127) / 128;   // 73

    // weight cast + transpose (bf16, [N][K])
    castT_kernel<<<1024, 256, 0, stream>>>(wq, wqT, NC, NC);
    castT_kernel<<<1024, 256, 0, stream>>>(wk, wkT, NC, NC);
    castT_kernel<<<1024, 256, 0, stream>>>(wv, wvT, NC, NC);
    castT_kernel<<<1024, 256, 0, stream>>>(wo, woT, NC, NC);
    castT_kernel<<<2048, 256, 0, stream>>>(w1, w1T, NC, NFF);
    castT_kernel<<<2048, 256, 0, stream>>>(w2, w2T, NFF, NC);

    // LN1 -> bf16
    ln_kernel<<<MROWS, 256, 0, stream>>>(x, ln1g, ln1b, hbuf);

    // QKV projections (head-split output layout)
    gemm_kernel<0><<<dim3(NC / 128, tilesM), 256, 0, stream>>>(hbuf, wqT, bq, nullptr, Qb, MROWS, NC, NC);
    gemm_kernel<0><<<dim3(NC / 128, tilesM), 256, 0, stream>>>(hbuf, wkT, bk, nullptr, Kb, MROWS, NC, NC);
    gemm_kernel<0><<<dim3(NC / 128, tilesM), 256, 0, stream>>>(hbuf, wvT, bv, nullptr, Vb, MROWS, NC, NC);

    // attention
    attn_kernel<<<dim3((NTOK + 127) / 128, NB * NH), 256, 0, stream>>>(Qb, Kb, Vb, AOb);

    // output projection + residual -> x1 (fp32)
    gemm_kernel<1><<<dim3(NC / 128, tilesM), 256, 0, stream>>>(AOb, woT, bo, x, x1, MROWS, NC, NC);

    // LN2 -> bf16
    ln_kernel<<<MROWS, 256, 0, stream>>>(x1, ln2g, ln2b, h2);

    // MLP up + exact GELU -> bf16
    gemm_kernel<2><<<dim3(NFF / 128, tilesM), 256, 0, stream>>>(h2, w1T, b1, nullptr, mid, MROWS, NC, NFF);

    // MLP down + residual -> d_out (fp32)
    gemm_kernel<1><<<dim3(NC / 128, tilesM), 256, 0, stream>>>(mid, w2T, b2, x1, (float*)d_out, MROWS, NFF, NC);
}
